// GraphAttentionTemplate_89172110999636
// MI455X (gfx1250) — matrix operand order repaired and checked by reading, not compiled, not run on hardware
//
#include <hip/hip_runtime.h>
#include <stdint.h>

// B=256, N=128, D=2048, DH=128. One workgroup per batch, fully fused:
//   h = X[b] @ Wa  via bf16 hi/lo split WMMA (3x v_wmma_f32_16x16x32_bf16, f32 acc)
//   t1/t2 projections, leaky_relu, softmax over axis=1, out = (coefs+1)*h + bias
// X chunk staged fp32 via global_load_async_to_lds_b128 (ASYNCcnt, NT hint);
// Wa chunk staged once as transposed, K-pair-packed bf16 hi/lo so B fragments
// are pure ds_load_b128 (no per-use conversion VALU).

typedef __attribute__((ext_vector_type(16))) __bf16    v16bf;
typedef __attribute__((ext_vector_type(8)))  float     v8f;
typedef __attribute__((ext_vector_type(4)))  unsigned  vu4;

#define N_   128
#define D_   2048
#define DH_  128
#define KC   128    // K-chunk staged in LDS
#define SX   132    // padded fp32 LDS row stride (floats), 16B-aligned rows
#define SKD  68     // padded packed-bf16 row stride (dwords = K-pairs), 16B-aligned

// LDS map (floats/dwords):
//   [0, 16896)        : X chunk fp32 [128][SX]; reused as h [128][SX] after GEMM
//   [16896, 25600)    : WtH packed bf16-hi, [n=128][SKD]
//   [25600, 34304)    : WtL packed bf16-lo, [n=128][SKD]
//   [33792, 34304)    : t1[128], t2[128], mj[128], iv[128] (reuses WtL tail after GEMM)
#define SMEM_FLOATS 34816
#define SMEM_BYTES  (SMEM_FLOATS * 4)   // 139264 B -> 2 blocks / 320KB WGP

__device__ __forceinline__ float lrelu(float x) { return x > 0.f ? x : 0.2f * x; }

__device__ __forceinline__ unsigned pack2(__bf16 a, __bf16 b) {
    return (unsigned)__builtin_bit_cast(unsigned short, a)
         | ((unsigned)__builtin_bit_cast(unsigned short, b) << 16);
}

__launch_bounds__(256)
__global__ void gat_fused_kernel(const float* __restrict__ X,
                                 const float* __restrict__ Wa,
                                 const float* __restrict__ Wb,
                                 const float* __restrict__ bb,
                                 const float* __restrict__ Wc,
                                 const float* __restrict__ bc,
                                 const float* __restrict__ bias,
                                 float* __restrict__ out) {
    extern __shared__ float smem[];
    float*    Xl  = smem;                        // [128][SX] fp32; later h
    unsigned* WtH = (unsigned*)(smem + 16896);   // [128][SKD] packed bf16 hi
    unsigned* WtL = WtH + 128 * SKD;             // [128][SKD] packed bf16 lo
    float*    hL  = smem;
    float*    t1  = smem + 34304 - 512;          // = smem + 33792
    float*    t2  = t1 + 128;
    float*    mj  = t2 + 128;
    float*    iv  = mj + 128;

    const int b    = blockIdx.x;
    const int tid  = threadIdx.x;
    const int wave = tid >> 5;     // wave w owns h rows 16w..16w+15
    const int lane = tid & 31;
    const int m16  = lane & 15;
    const int kh   = lane >> 4;    // half-wave selector in WMMA layouts

    const uint64_t xbase = (uint64_t)(uintptr_t)(X + (size_t)b * N_ * D_);

    v8f acc[8] = {};               // 8 N-tiles of 16x16 f32 accumulators

    const int wn   = tid & 127;    // column of Wa this thread stages
    const int whlf = tid >> 7;

    #pragma unroll 1
    for (int k0 = 0; k0 < D_; k0 += KC) {
        __syncthreads();  // previous chunk fully consumed before overwrite

        // --- async-stage X[b][:, k0:k0+128] fp32 into LDS (streamed once: NT) ---
        #pragma unroll
        for (int s = 0; s < 16; ++s) {
            int r = (tid >> 5) + 8 * s;              // row 0..127
            int g = tid & 31;                        // 4-float group
            unsigned ldsx  = (unsigned)(uintptr_t)(Xl + r * SX + 4 * g);
            unsigned voffx = (unsigned)((r * D_ + k0 + 4 * g) * 4);
            asm volatile("global_load_async_to_lds_b128 %0, %1, %2 th:TH_LOAD_NT"
                         :: "v"(ldsx), "v"(voffx), "s"(xbase) : "memory");
        }

        // --- stage Wa[k0:k0+128, :] as transposed packed bf16 hi/lo ---
        // thread (wn, whlf) covers kgroups kg = 8*whlf + it; 8 consecutive k each.
        #pragma unroll
        for (int it = 0; it < 8; ++it) {
            const int kg = 8 * whlf + it;            // 0..15
            const int kb = kg * 8;
            float x[8];
            #pragma unroll
            for (int k = 0; k < 8; ++k)
                x[k] = Wa[(size_t)(k0 + kb + k) * DH_ + wn];   // coalesced over wn
            vu4 hp, lp;
            #pragma unroll
            for (int v = 0; v < 4; ++v) {
                float x0 = x[2 * v], x1 = x[2 * v + 1];
                __bf16 h0 = (__bf16)x0, h1 = (__bf16)x1;
                __bf16 l0 = (__bf16)(x0 - (float)h0);
                __bf16 l1 = (__bf16)(x1 - (float)h1);
                hp[v] = pack2(h0, h1);               // low = even K
                lp[v] = pack2(l0, l1);
            }
            *(vu4*)(WtH + wn * SKD + kg * 4) = hp;   // 16B aligned
            *(vu4*)(WtL + wn * SKD + kg * 4) = lp;
        }

        asm volatile("s_wait_asynccnt 0" ::: "memory");
        __syncthreads();

        // --- 4 sub-steps of K=32 per chunk ---
        #pragma unroll
        for (int kk = 0; kk < 4; ++kk) {
            const int kb = kk * 32;
            // A fragment (16x32 bf16): lanes 0-15 K {0-7,16-23}, lanes 16-31 {8-15,24-31}
            v16bf a_hi, a_lo;
            {
                const float* ap = Xl + (wave * 16 + m16) * SX + kb + 8 * kh;
                float4 p0 = *(const float4*)(ap);
                float4 p1 = *(const float4*)(ap + 4);
                float4 q0 = *(const float4*)(ap + 16);
                float4 q1 = *(const float4*)(ap + 20);
                float va[16] = {p0.x, p0.y, p0.z, p0.w, p1.x, p1.y, p1.z, p1.w,
                                q0.x, q0.y, q0.z, q0.w, q1.x, q1.y, q1.z, q1.w};
                #pragma unroll
                for (int e = 0; e < 16; ++e) {
                    float xv = va[e];
                    __bf16 h_ = (__bf16)xv;
                    a_hi[e] = h_;
                    a_lo[e] = (__bf16)(xv - (float)h_);
                }
            }
            #pragma unroll
            for (int t = 0; t < 8; ++t) {
                // B fragment: lane -> col n, K {8kh..8kh+7, 16+8kh..16+8kh+7}
                // = packed dwords kd = 16*kk + 4*kh + {0..3, 8..11} of row n in Wt.
                const unsigned* bh = WtH + (t * 16 + m16) * SKD + kk * 16 + 4 * kh;
                const unsigned* bl = WtL + (t * 16 + m16) * SKD + kk * 16 + 4 * kh;
                union { vu4 q[2]; v16bf v; } ub, uc;
                ub.q[0] = *(const vu4*)(bh);
                ub.q[1] = *(const vu4*)(bh + 8);
                uc.q[0] = *(const vu4*)(bl);
                uc.q[1] = *(const vu4*)(bl + 8);
                v16bf b_hi = ub.v, b_lo = uc.v;
                acc[t] = __builtin_amdgcn_wmma_f32_16x16x32_bf16(
                             false, a_hi, false, b_hi, (short)0, acc[t], false, false);
                acc[t] = __builtin_amdgcn_wmma_f32_16x16x32_bf16(
                             false, a_hi, false, b_lo, (short)0, acc[t], false, false);
                acc[t] = __builtin_amdgcn_wmma_f32_16x16x32_bf16(
                             false, a_lo, false, b_hi, (short)0, acc[t], false, false);
            }
        }
    }
    __syncthreads();

    // Spill h to LDS. C/D layout: VGPR r -> M = r (lanes 0-15) / 8+r (lanes 16-31).
    #pragma unroll
    for (int t = 0; t < 8; ++t) {
        const int i0 = wave * 16 + kh * 8;
        const int j  = t * 16 + m16;
        #pragma unroll
        for (int r = 0; r < 8; ++r) hL[(i0 + r) * SX + j] = acc[t][r];
    }
    __syncthreads();

    // t1[i] = h[i,:].Wb + bb (threads 0-127); t2[i] = h[i,:].Wc + bc (threads 128-255)
    {
        const int i = tid & 127;
        const float* w = (tid < 128) ? Wb : Wc;
        float s = 0.f;
        #pragma unroll 4
        for (int j = 0; j < 128; ++j) s += hL[i * SX + j] * w[j];
        if (tid < 128) t1[i] = s + bb[0];
        else           t2[i] = s + bc[0];
    }
    __syncthreads();

    // Per-column (j) softmax stats over rows i of lrelu(t1[i] + t2[j]).
    if (tid < 128) {
        const int j = tid;
        const float tj = t2[j];
        float mx = -3.402823466e38f;
        for (int i = 0; i < 128; ++i) mx = fmaxf(mx, lrelu(t1[i] + tj));
        float sum = 0.f;
        for (int i = 0; i < 128; ++i) sum += __expf(lrelu(t1[i] + tj) - mx);
        mj[j] = mx;
        iv[j] = 1.f / sum;
    }
    __syncthreads();

    // out[i,j] = (coefs[i,j] + 1) * h[i,j] + bias[j], coalesced float4 stores.
    float* outb = out + (size_t)b * N_ * DH_;
    #pragma unroll
    for (int s = 0; s < 16; ++s) {
        int q  = tid + 256 * s;      // float4 id 0..4095
        int i  = q >> 5;
        int jg = (q & 31) * 4;
        float ti = t1[i];
        float4 r;
        float* rp = &r.x;
        #pragma unroll
        for (int c = 0; c < 4; ++c) {
            int j = jg + c;
            float cf = __expf(lrelu(ti + t2[j]) - mj[j]) * iv[j];
            rp[c] = (cf + 1.f) * hL[i * SX + j] + bias[j];
        }
        *(float4*)(outb + i * DH_ + jg) = r;
    }
}

extern "C" void kernel_launch(void* const* d_in, const int* in_sizes, int n_in,
                              void* d_out, int out_size, void* d_ws, size_t ws_size,
                              hipStream_t stream) {
    (void)in_sizes; (void)n_in; (void)d_ws; (void)ws_size; (void)out_size;
    const float* X    = (const float*)d_in[0];
    const float* Wa   = (const float*)d_in[1];
    const float* Wb   = (const float*)d_in[2];
    const float* bb   = (const float*)d_in[3];
    const float* Wc   = (const float*)d_in[4];
    const float* bc   = (const float*)d_in[5];
    const float* bias = (const float*)d_in[6];
    float* out = (float*)d_out;

    static_assert(SMEM_BYTES <= 320 * 1024, "LDS budget");
    hipFuncSetAttribute((const void*)gat_fused_kernel,
                        hipFuncAttributeMaxDynamicSharedMemorySize, SMEM_BYTES);
    gat_fused_kernel<<<dim3(256), dim3(256), SMEM_BYTES, stream>>>(
        X, Wa, Wb, bb, Wc, bc, bias, out);
}
